// RoPERotarySparseSelfAttention_74577812127950
// MI455X (gfx1250) — matrix operand order repaired and checked by reading, not compiled, not run on hardware
//
#include <hip/hip_runtime.h>
#include <hip/hip_bf16.h>
#include <math.h>

typedef __attribute__((ext_vector_type(16))) __bf16 v16bf;
typedef __attribute__((ext_vector_type(8)))  __bf16 v8bf;
typedef __attribute__((ext_vector_type(4)))  __bf16 v4bf;
typedef __attribute__((ext_vector_type(8)))  float  v8f;

#define B_    2
#define T_    2048
#define C_    2048
#define H_    16
#define D_    128
#define M_    (B_*T_)      /* 4096 rows (B*T)        */
#define N1_   (3*C_)       /* 6144 qkv cols          */
#define SPK   64           /* sparsity k             */

// ---------------------------------------------------------------------------
// 16-byte global -> LDS copy.  On gfx1250 use the ASYNCcnt-tracked direct
// path (no VGPR round-trip): GLOBAL_LOAD_ASYNC_TO_LDS_B128.  Generic LDS
// pointers carry the LDS byte offset in their low 32 bits (ISA 10.2 aperture
// rule), which is exactly what the VDST operand wants.
// ---------------------------------------------------------------------------
__device__ __forceinline__ void async_copy16(__bf16* ldsp, const __bf16* gp)
{
#if defined(__gfx1250__)
    unsigned lofs = (unsigned)(uintptr_t)ldsp;
    asm volatile("global_load_async_to_lds_b128 %0, %1, off"
                 :: "v"(lofs), "v"(gp) : "memory");
#else
    *(v8bf*)ldsp = *(const v8bf*)gp;
#endif
}

__device__ __forceinline__ void wait_async0()
{
#if defined(__gfx1250__)
    asm volatile("s_wait_asynccnt 0x0" ::: "memory");
#endif
}

// ---------------------------------------------------------------------------
// fp32 -> bf16 elementwise (vectorized x4)
// ---------------------------------------------------------------------------
__global__ __launch_bounds__(256)
void convert_f32_bf16_x4(const float4* __restrict__ in, v4bf* __restrict__ out, int n4)
{
    int i = blockIdx.x * blockDim.x + threadIdx.x;
    if (i < n4) {
        float4 v = in[i];
        v4bf o;
        o[0] = (__bf16)v.x; o[1] = (__bf16)v.y; o[2] = (__bf16)v.z; o[3] = (__bf16)v.w;
        out[i] = o;
    }
}

// ---------------------------------------------------------------------------
// W[K][N] fp32 -> Wt[N][K] bf16 (LDS-tiled transpose, 32x32 tile, 256 thr)
// ---------------------------------------------------------------------------
__global__ __launch_bounds__(256)
void transpose_f32_to_bf16(const float* __restrict__ W, __bf16* __restrict__ Wt,
                           int K, int N)
{
    __shared__ float tile[32][33];
    int n0 = blockIdx.x * 32, k0 = blockIdx.y * 32;
    int tx = threadIdx.x & 31, ty = threadIdx.x >> 5;   // 32 x 8
#pragma unroll
    for (int i = 0; i < 32; i += 8)
        tile[ty + i][tx] = W[(size_t)(k0 + ty + i) * N + n0 + tx];
    __syncthreads();
#pragma unroll
    for (int i = 0; i < 32; i += 8)
        Wt[(size_t)(n0 + ty + i) * K + k0 + tx] = (__bf16)tile[tx][ty + i];
}

// ---------------------------------------------------------------------------
// bf16 WMMA GEMM:  out[M][N] = A[M][K] * Bt[N][K]^T + bias[N]
// 256x128 block tile, 8 waves (wave32) in 4(M)x2(N) of 64x64 each (4x4 WMMA
// accumulators).  Double-buffered LDS staging via async global->LDS b128
// copies (ASYNCcnt) overlapped with WMMA on the other buffer.
// LDS tiles stride 40 bf16 (80B) -> 16B-aligned, bank-conflict-free b128 reads.
// Fragment layouts used here (contiguous k order, same for A and B):
//   A (16x32 bf16): lane<16 K{0..15}, lane>=16 K{16..31}, row=lane&15
//   B (32x16 bf16): lane<16 K{0..15}, lane>=16 K{16..31}, col=lane&15
// ---------------------------------------------------------------------------
template <typename OutT>
__global__ __launch_bounds__(256)
void gemm_bf16_wmma(const __bf16* __restrict__ A,
                    const __bf16* __restrict__ Bt,
                    const float*  __restrict__ bias,
                    OutT* __restrict__ out,
                    int M, int N, int K)
{
    __shared__ __bf16 lA[2][256 * 40];   // 2 x 20 KB
    __shared__ __bf16 lB[2][128 * 40];   // 2 x 10 KB
    const int tid   = threadIdx.x;
    const int lane  = tid & 31;
    const int wave  = tid >> 5;
    const int tileM = blockIdx.y * 256;
    const int tileN = blockIdx.x * 128;
    const int waveM = (wave >> 1) * 64;
    const int waveN = (wave & 1) * 64;
    const int rsel  = lane & 15;
    const int ahk   = (lane >> 4) * 16;    // A K-half (elements)
    const int bhk   = (lane >> 4) * 16;    // B K-half (elements)

    auto stage = [&](int buf, int k0) {
        // A: 256 rows x 32 cols = 1024 16B chunks; 256 threads -> 4 each
#pragma unroll
        for (int it = 0; it < 4; ++it) {
            int c = tid + it * 256;
            int row = c >> 2, seg = c & 3;
            async_copy16(&lA[buf][row * 40 + seg * 8],
                         &A[(size_t)(tileM + row) * K + k0 + seg * 8]);
        }
        // B: 128 rows -> 512 chunks -> 2 each
#pragma unroll
        for (int it = 0; it < 2; ++it) {
            int c = tid + it * 256;
            int row = c >> 2, seg = c & 3;
            async_copy16(&lB[buf][row * 40 + seg * 8],
                         &Bt[(size_t)(tileN + row) * K + k0 + seg * 8]);
        }
    };

    v8f acc[4][4];
#pragma unroll
    for (int ni = 0; ni < 4; ++ni) {
        float bv = bias[tileN + waveN + ni * 16 + rsel];
#pragma unroll
        for (int mi = 0; mi < 4; ++mi)
#pragma unroll
            for (int e = 0; e < 8; ++e) acc[mi][ni][e] = bv;
    }

    stage(0, 0);                         // prologue: fill buffer 0

    for (int k0 = 0; k0 < K; k0 += 32) {
        const int cur = (k0 >> 5) & 1;
        wait_async0();                   // this wave's outstanding async copies
        __syncthreads();                 // all waves: cur ready, cur^1 free
        if (k0 + 32 < K) stage(cur ^ 1, k0 + 32);   // overlap with compute

        v16bf aF[4], bF[4];
#pragma unroll
        for (int mi = 0; mi < 4; ++mi) {
            const __bf16* p = &lA[cur][(waveM + mi * 16 + rsel) * 40 + ahk];
            v8bf lo = *(const v8bf*)p;
            v8bf hi = *(const v8bf*)(p + 8);
#pragma unroll
            for (int e = 0; e < 8; ++e) { aF[mi][e] = lo[e]; aF[mi][e + 8] = hi[e]; }
        }
#pragma unroll
        for (int ni = 0; ni < 4; ++ni) {
            const __bf16* p = &lB[cur][(waveN + ni * 16 + rsel) * 40 + bhk];
            v8bf lo = *(const v8bf*)p;
            v8bf hi = *(const v8bf*)(p + 8);
#pragma unroll
            for (int e = 0; e < 8; ++e) { bF[ni][e] = lo[e]; bF[ni][e + 8] = hi[e]; }
        }
#pragma unroll
        for (int mi = 0; mi < 4; ++mi)
#pragma unroll
            for (int ni = 0; ni < 4; ++ni)
                acc[mi][ni] = __builtin_amdgcn_wmma_f32_16x16x32_bf16(
                    false, aF[mi], false, bF[ni], (short)0, acc[mi][ni], false, false);
    }

    const int mofs = (lane < 16) ? 0 : 8;
#pragma unroll
    for (int mi = 0; mi < 4; ++mi)
#pragma unroll
        for (int ni = 0; ni < 4; ++ni) {
            int n  = tileN + waveN + ni * 16 + rsel;
            int mb = tileM + waveM + mi * 16 + mofs;
#pragma unroll
            for (int e = 0; e < 8; ++e)
                out[(size_t)(mb + e) * N + n] = (OutT)acc[mi][ni][e];
        }
}

// ---------------------------------------------------------------------------
// RoPE in-place on bf16 qkv[4096][6144]: sections q (cols 0..2047) and
// k (2048..4095), per-head pairs (2i, 2i+1), D=128.
// ---------------------------------------------------------------------------
__global__ __launch_bounds__(256)
void rope_kernel(__bf16* __restrict__ qkv)
{
    int gid = blockIdx.x * blockDim.x + threadIdx.x;   // 4096*2048 threads
    int row = gid >> 11;          // / 2048 pairs-per-row
    int p   = gid & 2047;
    int sec = p >> 10;            // 0=q, 1=k
    int pi  = p & 1023;
    int h   = pi >> 6;
    int d2  = pi & 63;
    int col = sec * C_ + h * D_ + d2 * 2;
    int t   = row & (T_ - 1);
    float freq = __powf(10000.0f, -(float)(2 * d2) * (1.0f / 128.0f));
    float a = (float)t * freq;
    float s = __sinf(a), c = __cosf(a);
    size_t base = (size_t)row * N1_ + col;
    float xe = (float)qkv[base], xo = (float)qkv[base + 1];
    qkv[base]     = (__bf16)(xe * c - xo * s);
    qkv[base + 1] = (__bf16)(xe * s + xo * c);
}

// ---------------------------------------------------------------------------
// Per-(h,j) expiry: first time step t at which 64 prefix keys beat key j
// (strict total order: weight desc, index asc).  mask[h,t,j] <=> j<=t<e[h,j].
// ---------------------------------------------------------------------------
__global__ __launch_bounds__(256)
void expiry_kernel(const float* __restrict__ rw, int* __restrict__ expv)
{
    int gid = blockIdx.x * blockDim.x + threadIdx.x;   // H*T = 32768
    int h = gid >> 11, j = gid & (T_ - 1);
    float sj = rw[h * T_ + j];
    int cnt = 0, e = T_;
    for (int i = 0; i < T_; ++i) {
        float si = rw[h * T_ + i];
        bool beat = (si > sj) || (si == sj && i < j);
        cnt += beat;
        if (cnt == SPK) { e = i; break; }
    }
    expv[gid] = e;
}

// ---------------------------------------------------------------------------
// Wave32-ballot compaction: per (h,t), gather the <=64 active key indices.
// ---------------------------------------------------------------------------
__global__ __launch_bounds__(256)
void gather_topk_kernel(const int* __restrict__ expv,
                        int* __restrict__ idxb, int* __restrict__ cntb)
{
    const int w    = threadIdx.x >> 5;
    const int lane = threadIdx.x & 31;
    const int qid  = blockIdx.x * 8 + w;   // h*T + t
    const int t    = qid & (T_ - 1);
    const int* eh  = expv + (qid >> 11) * T_;
    int count = 0;
    for (int bj = 0; bj <= t; bj += 32) {
        int j = bj + lane;
        bool act = (j <= t) && (eh[j] > t);
        unsigned mask = (unsigned)__ballot(act);
        int pos = count + __popc(mask & ((1u << lane) - 1u));
        if (act && pos < SPK) idxb[qid * SPK + pos] = j;
        count += __popc(mask);
    }
    if (lane == 0) cntb[qid] = (count < SPK) ? count : SPK;
}

// ---------------------------------------------------------------------------
// Sparse attention: one wave32 per (b,h,t).  Scores: lane=key (2 keys/lane),
// shfl-xor softmax; PV: lane=4 output dims.  K/V stream hits L2 (64MB total).
// ---------------------------------------------------------------------------
__global__ __launch_bounds__(256)
void sparse_attn_kernel(const __bf16* __restrict__ qkv,
                        const int* __restrict__ idxb,
                        const int* __restrict__ cntb,
                        __bf16* __restrict__ aout)
{
    __shared__ float qs[8][128];
    __shared__ float ps[8][64];
    __shared__ int   js[8][64];
    const int w    = threadIdx.x >> 5;
    const int lane = threadIdx.x & 31;
    const int qid  = blockIdx.x * 8 + w;     // b*H*T + h*T + t
    const int b  = qid >> 15;
    const int ht = qid & 32767;
    const int h  = ht >> 11;
    const int t  = ht & (T_ - 1);
    const size_t row = (size_t)b * T_ + t;
    const __bf16* qrow = qkv + row * N1_ + h * D_;

#pragma unroll
    for (int i = 0; i < 4; ++i) qs[w][lane + 32 * i] = (float)qrow[lane + 32 * i];
    __syncthreads();

    const int cnt  = cntb[ht];
    const int base = ht * SPK;
    const float scale = 0.08838834764831845f;   // 1/sqrt(128)
    float s0 = -INFINITY, s1 = -INFINITY;
    int j0 = 0, j1 = 0;

    if (lane < cnt) {
        j0 = idxb[base + lane];
        const __bf16* kp = qkv + ((size_t)b * T_ + j0) * N1_ + C_ + h * D_;
        float a = 0.f;
        for (int it = 0; it < 16; ++it) {
            v8bf kk = *(const v8bf*)(kp + it * 8);
#pragma unroll
            for (int e = 0; e < 8; ++e) a += qs[w][it * 8 + e] * (float)kk[e];
        }
        s0 = a * scale;
    }
    if (lane + 32 < cnt) {
        j1 = idxb[base + lane + 32];
        const __bf16* kp = qkv + ((size_t)b * T_ + j1) * N1_ + C_ + h * D_;
        float a = 0.f;
        for (int it = 0; it < 16; ++it) {
            v8bf kk = *(const v8bf*)(kp + it * 8);
#pragma unroll
            for (int e = 0; e < 8; ++e) a += qs[w][it * 8 + e] * (float)kk[e];
        }
        s1 = a * scale;
    }

    float m = fmaxf(s0, s1);
#pragma unroll
    for (int off = 16; off > 0; off >>= 1) m = fmaxf(m, __shfl_xor(m, off, 32));
    float p0 = (lane < cnt)      ? __expf(s0 - m) : 0.f;
    float p1 = (lane + 32 < cnt) ? __expf(s1 - m) : 0.f;
    float l = p0 + p1;
#pragma unroll
    for (int off = 16; off > 0; off >>= 1) l += __shfl_xor(l, off, 32);
    float linv = 1.0f / l;

    ps[w][lane] = p0; ps[w][lane + 32] = p1;
    js[w][lane] = j0; js[w][lane + 32] = j1;
    __syncthreads();

    const int dbase = lane * 4;
    float o0 = 0, o1 = 0, o2 = 0, o3 = 0;
    for (int c = 0; c < cnt; ++c) {
        float p = ps[w][c];
        int   j = js[w][c];
        const __bf16* vp = qkv + ((size_t)b * T_ + j) * N1_ + 2 * C_ + h * D_ + dbase;
        v4bf vv = *(const v4bf*)vp;
        o0 += p * (float)vv[0]; o1 += p * (float)vv[1];
        o2 += p * (float)vv[2]; o3 += p * (float)vv[3];
    }
    __bf16* op = aout + row * C_ + h * D_ + dbase;
    op[0] = (__bf16)(o0 * linv); op[1] = (__bf16)(o1 * linv);
    op[2] = (__bf16)(o2 * linv); op[3] = (__bf16)(o3 * linv);
}

// ---------------------------------------------------------------------------
// Launch: all on `stream`; workspace-only scratch (no allocs).
// ---------------------------------------------------------------------------
extern "C" void kernel_launch(void* const* d_in, const int* in_sizes, int n_in,
                              void* d_out, int out_size, void* d_ws, size_t ws_size,
                              hipStream_t stream)
{
    (void)in_sizes; (void)n_in; (void)out_size; (void)ws_size;
    const float* x      = (const float*)d_in[0];
    const float* W_attn = (const float*)d_in[1];
    const float* b_attn = (const float*)d_in[2];
    const float* W_proj = (const float*)d_in[3];
    const float* b_proj = (const float*)d_in[4];
    const float* route  = (const float*)d_in[5];
    float* out = (float*)d_out;

    char* ws = (char*)d_ws;
    __bf16* xbf    = (__bf16*)ws;                 ws += (size_t)M_ * C_ * 2;     // 16 MB
    __bf16* wat    = (__bf16*)ws;                 ws += (size_t)N1_ * C_ * 2;    // 24 MB
    __bf16* wpt    = (__bf16*)ws;                 ws += (size_t)C_ * C_ * 2;     //  8 MB
    __bf16* qkv    = (__bf16*)ws;                 ws += (size_t)M_ * N1_ * 2;    // 48 MB
    __bf16* attnbf = (__bf16*)ws;                 ws += (size_t)M_ * C_ * 2;     // 16 MB
    int*    expv   = (int*)ws;                    ws += (size_t)H_ * T_ * 4;     // 128 KB
    int*    idxb   = (int*)ws;                    ws += (size_t)H_ * T_ * SPK*4; //  8 MB
    int*    cntb   = (int*)ws;

    // 1) x -> bf16
    {
        int n4 = (M_ * C_) / 4;
        convert_f32_bf16_x4<<<(n4 + 255) / 256, 256, 0, stream>>>(
            (const float4*)x, (v4bf*)xbf, n4);
    }
    // 2) W_attn^T, W_proj^T -> bf16
    transpose_f32_to_bf16<<<dim3(N1_ / 32, C_ / 32), 256, 0, stream>>>(W_attn, wat, C_, N1_);
    transpose_f32_to_bf16<<<dim3(C_ / 32,  C_ / 32), 256, 0, stream>>>(W_proj, wpt, C_, C_);
    // 3) qkv = x @ W_attn + b_attn   (bf16 out)   256x128 tiles
    gemm_bf16_wmma<__bf16><<<dim3(N1_ / 128, M_ / 256), 256, 0, stream>>>(
        xbf, wat, b_attn, qkv, M_, N1_, C_);
    // 4) RoPE on q,k
    rope_kernel<<<(M_ * 2048) / 256, 256, 0, stream>>>(qkv);
    // 5) top-64 mask via expiry + ballot gather
    expiry_kernel<<<(H_ * T_) / 256, 256, 0, stream>>>(route, expv);
    gather_topk_kernel<<<(H_ * T_) / 8, 256, 0, stream>>>(expv, idxb, cntb);
    // 6) sparse attention
    sparse_attn_kernel<<<(B_ * H_ * T_) / 8, 256, 0, stream>>>(qkv, idxb, cntb, attnbf);
    // 7) out = attn @ W_proj + b_proj  (fp32 out)
    gemm_bf16_wmma<float><<<dim3(C_ / 128, M_ / 256), 256, 0, stream>>>(
        attnbf, wpt, b_proj, out, M_, C_, C_);
}
